// CausalSelfAttention_1726576854189
// MI455X (gfx1250) — matrix operand order repaired and checked by reading, not compiled, not run on hardware
//
#include <hip/hip_runtime.h>
#include <math.h>

// ---------------------------------------------------------------------------
// CDNA5 (gfx1250) causal self-attention, bf16 WMMA pipeline.
// ---------------------------------------------------------------------------

typedef __bf16 bf16;
typedef __attribute__((ext_vector_type(4)))  __bf16 v4bf;
typedef __attribute__((ext_vector_type(8)))  __bf16 v8bf;
typedef __attribute__((ext_vector_type(16))) __bf16 v16bf;
typedef __attribute__((ext_vector_type(8)))  float  v8f;

#define EMBED 1024
#define HEADS 16
#define HDIM  64
#define CTX   2048
#define BATCH 4
#define NTOK  (BATCH * CTX)     // 8192

// -------------------- WMMA fragment helpers (wave32) -----------------------
// A-matrix 16x32 bf16 (M x K), row-major source with leading dim lda:
//   lanes 0-15  : row M=lane,   VGPR0-7 hold K=0..15
//   lanes 16-31 : row M=lane-16,VGPR0-7 hold K=16..31
__device__ inline v16bf load_a_frag(const bf16* base, int lda, int lane) {
  const int m  = lane & 15;
  const int hi = (lane >> 4) & 1;
  const bf16* p = base + (size_t)m * lda + hi * 16;
  union { v16bf v; v8bf h[2]; } u;
  u.h[0] = *(const v8bf*)(p);        // K = k0 + hi*16 .. +7   (16B)
  u.h[1] = *(const v8bf*)(p + 8);    // K = k0+hi*16+8 .. +15  (16B)
  return u.v;
}

// B-matrix 32x16 bf16 (K x N), where source is stored N-major (row n holds
// contiguous K): lanes 0-15 hold K=0..15 of column n=lane; lanes 16-31 hold
// K=16..31 of column n=lane-16.  One contiguous 32B load per lane.
__device__ inline v16bf load_b_frag(const bf16* base, int ldb, int lane) {
  const int n  = lane & 15;
  const int hi = (lane >> 4) & 1;
  return *(const v16bf*)(base + (size_t)n * ldb + hi * 16);
}

__device__ inline v8f wmma_bf16(v16bf a, v16bf b, v8f c) {
  return __builtin_amdgcn_wmma_f32_16x16x32_bf16(
      /*neg_a=*/false, a, /*neg_b=*/false, b,
      /*c_mod=*/(short)0, c, /*reuse_a=*/false, /*reuse_b=*/false);
}

// -------------------------- fp32 -> bf16 convert ---------------------------
__global__ __launch_bounds__(256) void cvt_f32_bf16(const float* __restrict__ in,
                                                    bf16* __restrict__ out, int n4) {
  int i = blockIdx.x * blockDim.x + threadIdx.x;
  if (i < n4) {
    float4 f = ((const float4*)in)[i];
    v4bf o;
    o.x = (bf16)f.x; o.y = (bf16)f.y; o.z = (bf16)f.z; o.w = (bf16)f.w;
    ((v4bf*)out)[i] = o;
  }
}

// ------------------------------- WMMA GEMM ---------------------------------
// y(M x N) = A(M x K, bf16 row-major) * W(N x K, bf16 row-major)^T + bias
// One wave computes a 32(M) x 64(N) tile: two A-fragments share the four
// B-fragments each K-step -> 8 WMMAs per 12 b128 loads.
// mode 0: fp32 output, row-major MxN (final projection -> d_out)
// mode 1: bf16 output, (B,H,T,D) layout, value scaled by `scale` (Q, K)
// mode 2: bf16 output, (B,H,D,T) layout (V stored transposed for P*V)
__global__ __launch_bounds__(128) void wmma_gemm(const bf16* __restrict__ A,
                                                 const bf16* __restrict__ W,
                                                 const float* __restrict__ bias,
                                                 void* __restrict__ out,
                                                 int M, int N, int K,
                                                 int mode, float scale) {
  const int lane = threadIdx.x & 31;
  const int wid  = blockIdx.x * (blockDim.x >> 5) + (threadIdx.x >> 5);
  const int n64  = N >> 6;
  const int mtile = wid / n64;
  const int ntile = wid % n64;
  if (mtile >= (M >> 5)) return;
  const int m0 = mtile << 5;           // 32 rows per wave
  const int n0 = ntile << 6;           // 64 cols per wave
  const int hi = (lane >> 4) & 1;
  const int lm = lane & 15;

  v8f acc[2][4] = {{v8f{}, v8f{}, v8f{}, v8f{}}, {v8f{}, v8f{}, v8f{}, v8f{}}};
  const bf16* Arow0 = A + (size_t)m0 * K;
  const bf16* Arow1 = A + (size_t)(m0 + 16) * K;
  for (int k0 = 0; k0 < K; k0 += 32) {
    const v16bf a0 = load_a_frag(Arow0 + k0, K, lane);
    const v16bf a1 = load_a_frag(Arow1 + k0, K, lane);
#pragma unroll
    for (int t = 0; t < 4; ++t) {
      const v16bf b = load_b_frag(W + (size_t)(n0 + t * 16) * K + k0, K, lane);
      acc[0][t] = wmma_bf16(a0, b, acc[0][t]);
      acc[1][t] = wmma_bf16(a1, b, acc[1][t]);
    }
  }

  // Epilogue.  C/D layout: lane (group of 16) holds column n = n0+t*16+lm;
  // VGPR r holds row m = m0 + mt*16 + r + 8*hi.
#pragma unroll
  for (int mt = 0; mt < 2; ++mt) {
#pragma unroll
    for (int t = 0; t < 4; ++t) {
#pragma unroll
      for (int r = 0; r < 8; ++r) {
        const int n = n0 + t * 16 + lm;
        const int m = m0 + mt * 16 + r + 8 * hi;
        const float v = acc[mt][t][r] + bias[n];
        if (mode == 0) {
          ((float*)out)[(size_t)m * N + n] = v;
        } else {
          const int bi = m >> 11;        // token -> batch
          const int tk = m & (CTX - 1);  // token within sequence
          const int hh = n >> 6;         // head
          const int dd = n & (HDIM - 1); // dim within head
          bf16* o = (bf16*)out;
          if (mode == 1) {
            o[(((size_t)(bi * HEADS + hh)) * CTX + tk) * HDIM + dd] = (bf16)(v * scale);
          } else { // mode 2: transposed V
            o[(((size_t)(bi * HEADS + hh)) * HDIM + dd) * CTX + tk] = (bf16)v;
          }
        }
      }
    }
  }
}

// ------------------------------ attention ----------------------------------
// One wave handles one (b, h, 16-query tile).  Flash-style online softmax
// over 32-key blocks.  Q is pre-scaled by 1/sqrt(D).
#define ATTN_WPB 8
__global__ __launch_bounds__(256) void attn_kernel(const bf16* __restrict__ Q,
                                                   const bf16* __restrict__ Km,
                                                   const bf16* __restrict__ Vt,
                                                   bf16* __restrict__ Oout) {
  __shared__ __align__(32) bf16 ldsP[ATTN_WPB][16 * 32];

  const int lane = threadIdx.x & 31;
  const int w    = threadIdx.x >> 5;
  const int wid  = blockIdx.x * ATTN_WPB + w;
  const int b  = wid >> 11;          // / (HEADS * CTX/16)
  const int h  = (wid >> 7) & 15;
  const int qt = wid & 127;
  const int m0 = qt << 4;
  const int hi = (lane >> 4) & 1;
  const int lm = lane & 15;

  const bf16* Qb = Q  + ((size_t)(b * HEADS + h) * CTX + m0) * HDIM;
  const bf16* Kb = Km + (size_t)(b * HEADS + h) * CTX * HDIM;
  const bf16* Vb = Vt + (size_t)(b * HEADS + h) * HDIM * CTX;

  // Q fragments: 16x64 split into two 16x32 A-fragments.
  const v16bf aq0 = load_a_frag(Qb + 0,  HDIM, lane);
  const v16bf aq1 = load_a_frag(Qb + 32, HDIM, lane);

  v8f oacc[4] = {v8f{}, v8f{}, v8f{}, v8f{}};
  float rowmax[8], rowsum[8];
#pragma unroll
  for (int r = 0; r < 8; ++r) { rowmax[r] = -__builtin_inff(); rowsum[r] = 0.0f; }

  const int nkb = (m0 + 16 + 31) >> 5;   // 32-key blocks covering keys <= m0+15
  for (int j = 0; j < nkb; ++j) {
    const int kb = j << 5;

    // S = Q * K^T for 32 keys -> two 16x16 f32 tiles.
    v8f c0 = v8f{}, c1 = v8f{};
    {
      v16bf bk;
      bk = load_b_frag(Kb + (size_t)kb * HDIM + 0,         HDIM, lane);
      c0 = wmma_bf16(aq0, bk, c0);
      bk = load_b_frag(Kb + (size_t)kb * HDIM + 32,        HDIM, lane);
      c0 = wmma_bf16(aq1, bk, c0);
      bk = load_b_frag(Kb + (size_t)(kb + 16) * HDIM + 0,  HDIM, lane);
      c1 = wmma_bf16(aq0, bk, c1);
      bk = load_b_frag(Kb + (size_t)(kb + 16) * HDIM + 32, HDIM, lane);
      c1 = wmma_bf16(aq1, bk, c1);
    }

    // Causal mask (only diagonal blocks need it).
    if (kb + 31 > m0) {
      const int k0c = kb + lm;
      const int k1c = kb + 16 + lm;
#pragma unroll
      for (int r = 0; r < 8; ++r) {
        const int q = m0 + r + 8 * hi;
        if (k0c > q) c0[r] = -__builtin_inff();
        if (k1c > q) c1[r] = -__builtin_inff();
      }
    }

    // Online softmax: rows are striped across the 16-lane group.
    float corr[8];
#pragma unroll
    for (int r = 0; r < 8; ++r) {
      float loc = fmaxf(c0[r], c1[r]);
#pragma unroll
      for (int off = 8; off > 0; off >>= 1)
        loc = fmaxf(loc, __shfl_xor(loc, off, 32));
      const float mnew = fmaxf(rowmax[r], loc);
      const float cr   = __expf(rowmax[r] - mnew);
      rowmax[r] = mnew;
      const float p0 = __expf(c0[r] - mnew);
      const float p1 = __expf(c1[r] - mnew);
      float psum = p0 + p1;
#pragma unroll
      for (int off = 8; off > 0; off >>= 1)
        psum += __shfl_xor(psum, off, 32);
      rowsum[r] = rowsum[r] * cr + psum;
      corr[r] = cr;
      // Stage P (bf16) into LDS row-major 16x32 for the A-fragment reload.
      const int row = r + 8 * hi;
      ldsP[w][row * 32 + lm]      = (bf16)p0;
      ldsP[w][row * 32 + 16 + lm] = (bf16)p1;
    }
#pragma unroll
    for (int t = 0; t < 4; ++t)
#pragma unroll
      for (int r = 0; r < 8; ++r) oacc[t][r] *= corr[r];

    // Same-wave LDS ops are in-order; wait for writes before fragment reads.
    __asm volatile("s_wait_dscnt 0" ::: "memory");

    // P (16x32) x V (32x64) via pre-transposed V: contiguous B fragments.
    const v16bf ap = load_a_frag(&ldsP[w][0], 32, lane);
#pragma unroll
    for (int t = 0; t < 4; ++t) {
      v16bf bv = load_b_frag(Vb + (size_t)(t * 16) * CTX + kb, CTX, lane);
      oacc[t] = wmma_bf16(ap, bv, oacc[t]);
    }
  }

  // Normalize and store to (B,T,E) bf16 for the output projection.
#pragma unroll
  for (int t = 0; t < 4; ++t) {
#pragma unroll
    for (int r = 0; r < 8; ++r) {
      const float v = oacc[t][r] / rowsum[r];
      const int row = m0 + r + 8 * hi;
      const int col = h * HDIM + t * 16 + lm;
      Oout[((size_t)b * CTX + row) * EMBED + col] = (bf16)v;
    }
  }
}

// ------------------------------ launcher -----------------------------------
extern "C" void kernel_launch(void* const* d_in, const int* in_sizes, int n_in,
                              void* d_out, int out_size, void* d_ws, size_t ws_size,
                              hipStream_t stream) {
  (void)in_sizes; (void)n_in; (void)out_size; (void)ws_size;
  const float* x  = (const float*)d_in[0];
  const float* Wq = (const float*)d_in[1];
  const float* bq = (const float*)d_in[2];
  const float* Wk = (const float*)d_in[3];
  const float* bk = (const float*)d_in[4];
  const float* Wv = (const float*)d_in[5];
  const float* bv = (const float*)d_in[6];
  const float* Wp = (const float*)d_in[7];
  const float* bp = (const float*)d_in[8];

  char* ws = (char*)d_ws;
  const size_t MB = 1024 * 1024;
  bf16* xb  = (bf16*)(ws + 0);        // 16 MB: x as bf16 (8192 x 1024)
  bf16* wqb = (bf16*)(ws + 16 * MB);  //  2 MB
  bf16* wkb = (bf16*)(ws + 18 * MB);
  bf16* wvb = (bf16*)(ws + 20 * MB);
  bf16* wpb = (bf16*)(ws + 22 * MB);
  bf16* Qs  = (bf16*)(ws + 24 * MB);  // 16 MB: Q  (B,H,T,D), pre-scaled
  bf16* Ks  = (bf16*)(ws + 40 * MB);  // 16 MB: K  (B,H,T,D)
  bf16* Vts = (bf16*)(ws + 56 * MB);  // 16 MB: V^T(B,H,D,T)
  bf16* AO  = (bf16*)(ws + 72 * MB);  // 16 MB: attention out (B,T,E)

  // 1) fp32 -> bf16 conversions.
  {
    const int nx4 = NTOK * EMBED / 4;
    const int nw4 = EMBED * EMBED / 4;
    cvt_f32_bf16<<<(nx4 + 255) / 256, 256, 0, stream>>>(x,  xb,  nx4);
    cvt_f32_bf16<<<(nw4 + 255) / 256, 256, 0, stream>>>(Wq, wqb, nw4);
    cvt_f32_bf16<<<(nw4 + 255) / 256, 256, 0, stream>>>(Wk, wkb, nw4);
    cvt_f32_bf16<<<(nw4 + 255) / 256, 256, 0, stream>>>(Wv, wvb, nw4);
    cvt_f32_bf16<<<(nw4 + 255) / 256, 256, 0, stream>>>(Wp, wpb, nw4);
  }

  // 2) QKV projections (WMMA).  waves = (M/32)*(N/64) = 4096 -> 1024 blocks.
  {
    const int blocks = (NTOK / 32) * (EMBED / 64) / 4;
    const float qscale = 1.0f / 8.0f;  // 1/sqrt(HDIM)
    wmma_gemm<<<blocks, 128, 0, stream>>>(xb, wqb, bq, Qs,  NTOK, EMBED, EMBED, 1, qscale);
    wmma_gemm<<<blocks, 128, 0, stream>>>(xb, wkb, bk, Ks,  NTOK, EMBED, EMBED, 1, 1.0f);
    wmma_gemm<<<blocks, 128, 0, stream>>>(xb, wvb, bv, Vts, NTOK, EMBED, EMBED, 2, 1.0f);
  }

  // 3) causal attention: 4*16*128 = 8192 waves, 8 waves/block.
  attn_kernel<<<(BATCH * HEADS * (CTX / 16)) / ATTN_WPB, 32 * ATTN_WPB, 0, stream>>>(
      Qs, Ks, Vts, AO);

  // 4) output projection -> fp32 d_out.
  {
    const int blocks = (NTOK / 32) * (EMBED / 64) / 4;
    wmma_gemm<<<blocks, 128, 0, stream>>>(AO, wpb, bp, d_out, NTOK, EMBED, EMBED, 0, 0.0f);
  }
}
